// GATModule_60576218743177
// MI455X (gfx1250) — matrix operand order repaired and checked by reading, not compiled, not run on hardware
//
#include <hip/hip_runtime.h>
#include <hip/hip_bf16.h>

// ---------------- problem constants ----------------
constexpr int Nn     = 6144;
constexpr int IN_F   = 256;
constexpr int OUT_F  = 128;
constexpr int HEADS  = 4;
constexpr int CH     = HEADS * OUT_F;   // 512
constexpr int KB_X   = IN_F / 32;       // 8  k-blocks in projection
constexpr int OT_N   = OUT_F / 16;      // 8  o-tiles
constexpr int JB_N   = Nn / 32;         // 192 j-blocks
constexpr float NEG_SLOPE = 0.2f;
constexpr float BN_EPS    = 1e-5f;
constexpr float L2E       = 1.4426950408889634f;   // log2(e)

typedef __attribute__((ext_vector_type(16))) _Float16 v16h;
typedef __attribute__((ext_vector_type(8)))  float    v8f;

// ======================================================================
// kernel 0a: swizzle x (f32 row-major) -> xA (f16, WMMA A-fragment order)
// A-frag: value(row=ln, k) at lane = ln + 16*((k>>3)&1), elem e=(k&7)+8*(k>>4)
// flat: ((i_tile*KB_X + kb)*32 + lane)*16 + e
// ======================================================================
__global__ void swz_x(const float* __restrict__ x, _Float16* __restrict__ xA) {
    int idx = blockIdx.x * blockDim.x + threadIdx.x;
    if (idx >= Nn * IN_F) return;
    int n = idx / IN_F, k = idx - n * IN_F;
    int it = n >> 4, ln = n & 15;
    int kb = k >> 5, kr = k & 31;
    int half = (kr >> 3) & 1;
    int e    = (kr & 7) + ((kr >> 4) << 3);
    int lane = ln + (half << 4);
    xA[(((size_t)it * KB_X + kb) * 32 + lane) * 16 + e] = (_Float16)x[idx];
}

// ======================================================================
// kernel 0b: swizzle W -> wB (f16, WMMA B-fragment order)
// B-frag: value(k, col=n) at lane = n + 16*((k>>3)&1), elem e=(k&7)+8*(k>>4)
// ======================================================================
__global__ void swz_w(const float* __restrict__ W, _Float16* __restrict__ wB) {
    int idx = blockIdx.x * blockDim.x + threadIdx.x;
    if (idx >= HEADS * IN_F * OUT_F) return;
    int head = idx / (IN_F * OUT_F);
    int rem  = idx - head * (IN_F * OUT_F);
    int k = rem / OUT_F, o = rem - k * OUT_F;
    int kb = k >> 5, kr = k & 31;
    int half = (kr >> 3) & 1, e = (kr & 7) + ((kr >> 4) << 3);
    int ot = o >> 4, n = o & 15;
    int lane = n + (half << 4);
    wB[((((size_t)head * KB_X + kb) * OT_N + ot) * 32 + lane) * 16 + e] = (_Float16)W[idx];
}

// ======================================================================
// kernel 1: projection  h = x @ W  (per head), WMMA f16->f32.
// Outputs row-major h16 (for attvec) and B-fragment-ordered hB (for flash).
// grid (Nn/16, OT_N, HEADS), block 32
// ======================================================================
__global__ void proj_wmma(const _Float16* __restrict__ xA,
                          const _Float16* __restrict__ wB,
                          _Float16* __restrict__ h16,
                          _Float16* __restrict__ hB)
{
    const int lane = threadIdx.x;
    const int half = lane >> 4;
    const int ln   = lane & 15;
    const int it   = blockIdx.x;
    const int ot   = blockIdx.y;
    const int head = blockIdx.z;

    const _Float16* ap = xA + ((size_t)it * KB_X) * 512 + lane * 16;
    const _Float16* bp = wB + (((size_t)head * KB_X) * OT_N + ot) * 512 + lane * 16;

    v8f c = {};
#pragma unroll
    for (int kb = 0; kb < KB_X; ++kb) {
        v16h a = *(const v16h*)(ap + (size_t)kb * 512);
        v16h b = *(const v16h*)(bp + (size_t)kb * OT_N * 512);
        c = __builtin_amdgcn_wmma_f32_16x16x32_f16(false, a, false, b, (short)0, c, false, false);
    }

    const int i0 = it * 16;
    const int jb = i0 >> 5;
    const int rbase = i0 & 31;
#pragma unroll
    for (int r = 0; r < 8; ++r) {
        int m = r + 8 * half;
        _Float16 v = (_Float16)c[r];
        h16[((size_t)head * Nn + (i0 + m)) * OUT_F + ot * 16 + ln] = v;
        int rb    = rbase + m;
        int halfB = rb >> 4, eB = rb & 15;
        int laneB = ln + (halfB << 4);
        hB[((((size_t)head * JB_N + jb) * OT_N + ot) * 32 + laneB) * 16 + eB] = v;
    }
}

// ======================================================================
// kernel 2: attention vectors ai/aj, pre-scaled by log2(e) for exp2-softmax
// (leaky_relu commutes with the positive scale L2E)
// ======================================================================
__global__ void attvec(const _Float16* __restrict__ h16,
                       const float* __restrict__ a_i, const float* __restrict__ a_j,
                       float* __restrict__ ai, float* __restrict__ aj)
{
    int idx = blockIdx.x * blockDim.x + threadIdx.x;  // over HEADS*Nn
    if (idx >= HEADS * Nn) return;
    int head = idx / Nn, n = idx - head * Nn;
    const _Float16* hp = h16 + ((size_t)head * Nn + n) * OUT_F;
    const float* vi = a_i + head * OUT_F;
    const float* vj = a_j + head * OUT_F;
    float si = 0.f, sj = 0.f;
#pragma unroll 4
    for (int o = 0; o < OUT_F; ++o) {
        float hv = (float)hp[o];
        si += hv * vi[o];
        sj += hv * vj[o];
    }
    ai[idx] = si * L2E; aj[idx] = sj * L2E;
}

// ======================================================================
// kernel 3: flash attention.
// block = 256 threads = 8 waves; wave w: head = w>>1, jpar = w&1.
// Each wave: 32-row i-tile (two 16-row sub-tiles), j-blocks j=(2p+jpar)*32.
// Double-buffered LDS staging of the (head-shared) mask tile: one barrier
// per iteration, next tile's global loads overlap current tile's compute.
// ======================================================================
__global__ void __launch_bounds__(256) flash_att(
    const float*     __restrict__ graph,   // [Nn][Nn] mask (0 or -1e9)
    const float*     __restrict__ aiv,     // [H][Nn]  (x log2e)
    const float*     __restrict__ ajv,     // [H][Nn]  (x log2e)
    const _Float16*  __restrict__ hB,      // B-fragment ordered h
    float*           __restrict__ y)       // [Nn][CH] pre-BN
{
    __shared__ float s_mask[2][32][68];            // ping-pong 32x64 tiles, stride 68
    __shared__ float s_aj[2][HEADS][64];
    __shared__ float s_acc[HEADS][8][16][16];      // merge buffer (one sub-tile at a time)
    __shared__ float s_ml[HEADS][2][16];           // merge stats: m,l per row

    const int tid  = threadIdx.x;
    const int wave = tid >> 5;
    const int head = wave >> 1;
    const int jpar = wave & 1;
    const int lane = tid & 31;
    const int half = lane >> 4;
    const int ln   = lane & 15;
    const int i0   = blockIdx.x * 32;

    // staging roles (fixed per thread)
    const int r_st  = tid >> 4;            // 0..15
    const int c4_st = (tid & 15) * 4;      // 0..60
    const int hh_st = tid >> 6, cc_st = tid & 63;

    const float ai0 = aiv[head * Nn + i0 + ln];
    const float ai1 = aiv[head * Nn + i0 + 16 + ln];

    v8f acc0[8], acc1[8];
#pragma unroll
    for (int ot = 0; ot < 8; ++ot) { acc0[ot] = (v8f){}; acc1[ot] = (v8f){}; }
    float m0 = -1e30f, l0 = 0.f, m1 = -1e30f, l1 = 0.f;

    const _Float16* hBh = hB + (size_t)head * JB_N * 4096;
    const int jco = jpar * 32;

    auto load_regs = [&](int p, float4& ga, float4& gb, float& aje) {
        const int jc0 = p * 64;
        ga  = *(const float4*)(graph + (size_t)(i0 + r_st) * Nn + jc0 + c4_st);
        gb  = *(const float4*)(graph + (size_t)(i0 + 16 + r_st) * Nn + jc0 + c4_st);
        aje = ajv[hh_st * Nn + jc0 + cc_st];
    };
    auto store_buf = [&](int buf, const float4& ga, const float4& gb, float aje) {
        // fold log2(e) into the mask during staging
        float4 sa = make_float4(ga.x * L2E, ga.y * L2E, ga.z * L2E, ga.w * L2E);
        float4 sb = make_float4(gb.x * L2E, gb.y * L2E, gb.z * L2E, gb.w * L2E);
        *(float4*)&s_mask[buf][r_st][c4_st]      = sa;
        *(float4*)&s_mask[buf][16 + r_st][c4_st] = sb;
        s_aj[buf][hh_st][cc_st] = aje;
    };

    // online-softmax step for one 16-row sub-tile (log2 domain, exp2)
    auto softstep = [&](int buf, float ai_r, float& m_cur, float& l_cur,
                        int trow, v16h& a) -> float {
        float ev[16];
        float lmax = -1e30f;
#pragma unroll
        for (int e = 0; e < 16; ++e) {
            int k = e + 16 * half;
            float t = ai_r + s_aj[buf][head][jco + k];
            t = (t >= 0.f) ? t : NEG_SLOPE * t;              // leaky_relu
            t += s_mask[buf][16 * trow + ln][jco + k];       // mask (x log2e)
            ev[e] = t;
            lmax = fmaxf(lmax, t);
        }
        float rmax  = fmaxf(lmax, __shfl_xor(lmax, 16));
        float m_new = fmaxf(m_cur, rmax);
        float scale = exp2f(m_cur - m_new);
        float lsum  = 0.f;
#pragma unroll
        for (int e = 0; e < 16; ++e) {
            float p = exp2f(ev[e] - m_new);
            lsum += p;
            a[e] = (_Float16)p;
        }
        float rsum = lsum + __shfl_xor(lsum, 16);
        l_cur = l_cur * scale + rsum;
        m_cur = m_new;
        return scale;
    };

    const int npair = Nn / 64;
    {   // prologue: stage tile 0
        float4 ga, gb; float aje;
        load_regs(0, ga, gb, aje);
        store_buf(0, ga, gb, aje);
    }

    for (int p = 0; p < npair; ++p) {
        const int buf = p & 1;
        __syncthreads();                            // tile p visible to everyone

        float4 nga, ngb; float naje;
        const bool more = (p + 1 < npair);
        if (more) load_regs(p + 1, nga, ngb, naje); // overlap with compute below
        if (p + 2 < npair) {                        // prefetch tile p+2 (global_prefetch_b8)
            const int jc2 = (p + 2) * 64;
            __builtin_prefetch(graph + (size_t)(i0 + r_st) * Nn + jc2 + c4_st, 0, 3);
            __builtin_prefetch(graph + (size_t)(i0 + 16 + r_st) * Nn + jc2 + c4_st, 0, 3);
        }

        v16h a0, a1;
        float sc0 = softstep(buf, ai0, m0, l0, 0, a0);
        float sc1 = softstep(buf, ai1, m1, l1, 1, a1);

        if (__any(sc0 < 1.0f)) {                    // rescale only when running max moved
#pragma unroll
            for (int r = 0; r < 8; ++r) {
                float c0 = __shfl(sc0, 8 * half + r);
#pragma unroll
                for (int ot = 0; ot < 8; ++ot) acc0[ot][r] *= c0;
            }
        }
        if (__any(sc1 < 1.0f)) {
#pragma unroll
            for (int r = 0; r < 8; ++r) {
                float c1 = __shfl(sc1, 8 * half + r);
#pragma unroll
                for (int ot = 0; ot < 8; ++ot) acc1[ot][r] *= c1;
            }
        }

        // B fragments: contiguous 32B per lane from fragment-ordered hB (L2 resident)
        const _Float16* bp = hBh + (size_t)(2 * p + jpar) * 4096 + lane * 16;
#pragma unroll
        for (int ot = 0; ot < 8; ++ot) {
            v16h b = *(const v16h*)(bp + ot * 512);
            acc0[ot] = __builtin_amdgcn_wmma_f32_16x16x32_f16(false, a0, false, b,
                                                              (short)0, acc0[ot], false, false);
            acc1[ot] = __builtin_amdgcn_wmma_f32_16x16x32_f16(false, a1, false, b,
                                                              (short)0, acc1[ot], false, false);
        }

        if (more) store_buf(1 - buf, nga, ngb, naje);   // safe: everyone passed this p's barrier
    }

    // ---- merge the two j-partitions per head, one sub-tile at a time ----
    auto merge = [&](v8f (&accT)[8], float mcur, float lcur, int t) {
        if (jpar == 1) {                 // partition B publishes state
#pragma unroll
            for (int ot = 0; ot < 8; ++ot)
#pragma unroll
                for (int r = 0; r < 8; ++r)
                    s_acc[head][ot][8 * half + r][ln] = accT[ot][r];
            if (lane < 16) { s_ml[head][0][ln] = mcur; s_ml[head][1][ln] = lcur; }
        }
        __syncthreads();
        if (jpar == 0) {                 // partition A merges + writes output
            float mB = s_ml[head][0][ln];
            float lB = s_ml[head][1][ln];
            float mM = fmaxf(mcur, mB);
            float cA = exp2f(mcur - mM);
            float cB = exp2f(mB - mM);
            float inv = 1.0f / (lcur * cA + lB * cB);   // self-loop => > 0
            float fA = cA * inv, fB = cB * inv;
#pragma unroll
            for (int r = 0; r < 8; ++r) {
                float fAr = __shfl(fA, 8 * half + r);
                float fBr = __shfl(fB, 8 * half + r);
                int m = 8 * half + r;
                float* yp = y + (size_t)(i0 + 16 * t + m) * CH + head * OUT_F;
#pragma unroll
                for (int ot = 0; ot < 8; ++ot)
                    yp[ot * 16 + ln] = accT[ot][r] * fAr + s_acc[head][ot][m][ln] * fBr;
            }
        }
        __syncthreads();
    };
    merge(acc0, m0, l0, 0);
    merge(acc1, m1, l1, 1);
}

// ======================================================================
// kernel 4: BatchNorm batch statistics (per channel over Nn rows)
// ======================================================================
__global__ void bn_stats(const float* __restrict__ y, float* __restrict__ mu, float* __restrict__ rs)
{
    const int c   = blockIdx.x;
    const int tid = threadIdx.x;
    float s = 0.f, s2 = 0.f;
    for (int n = tid; n < Nn; n += 256) {
        float v = y[(size_t)n * CH + c];
        s += v; s2 += v * v;
    }
    __shared__ float sh[256], sh2[256];
    sh[tid] = s; sh2[tid] = s2;
    __syncthreads();
    for (int st = 128; st > 0; st >>= 1) {
        if (tid < st) { sh[tid] += sh[tid + st]; sh2[tid] += sh2[tid + st]; }
        __syncthreads();
    }
    if (tid == 0) {
        float mean = sh[0] / (float)Nn;
        float var  = sh2[0] / (float)Nn - mean * mean;
        mu[c] = mean;
        rs[c] = rsqrtf(var + BN_EPS);
    }
}

// ======================================================================
// kernel 5: BN apply + ReLU (in place on d_out)
// ======================================================================
__global__ void bn_apply(float* __restrict__ y,
                         const float* __restrict__ mu, const float* __restrict__ rs,
                         const float* __restrict__ gamma, const float* __restrict__ beta)
{
    int idx = blockIdx.x * blockDim.x + threadIdx.x;
    if (idx >= Nn * CH) return;
    int c = idx & (CH - 1);
    float v = gamma[c] * (y[idx] - mu[c]) * rs[c] + beta[c];
    y[idx] = fmaxf(v, 0.f);
}

// ======================================================================
// launch
// ======================================================================
extern "C" void kernel_launch(void* const* d_in, const int* in_sizes, int n_in,
                              void* d_out, int out_size, void* d_ws, size_t ws_size,
                              hipStream_t stream)
{
    const float* x     = (const float*)d_in[0];
    const float* graph = (const float*)d_in[1];
    const float* W     = (const float*)d_in[2];
    const float* a_i   = (const float*)d_in[3];
    const float* a_j   = (const float*)d_in[4];
    const float* gamma = (const float*)d_in[5];
    const float* beta  = (const float*)d_in[6];
    float* y = (float*)d_out;

    char* base = (char*)d_ws;
    size_t off = 0;
    auto take = [&](size_t bytes) {
        char* p = base + off;
        off = (off + bytes + 255) & ~(size_t)255;
        return p;
    };
    _Float16* xA  = (_Float16*)take((size_t)Nn * IN_F * sizeof(_Float16));
    _Float16* wB  = (_Float16*)take((size_t)HEADS * IN_F * OUT_F * sizeof(_Float16));
    _Float16* h16 = (_Float16*)take((size_t)HEADS * Nn * OUT_F * sizeof(_Float16));
    _Float16* hB  = (_Float16*)take((size_t)HEADS * Nn * OUT_F * sizeof(_Float16));
    float*    ai  = (float*)take((size_t)HEADS * Nn * sizeof(float));
    float*    aj  = (float*)take((size_t)HEADS * Nn * sizeof(float));
    float*    mu  = (float*)take((size_t)CH * sizeof(float));
    float*    rs  = (float*)take((size_t)CH * sizeof(float));
    (void)ws_size; (void)in_sizes; (void)n_in; (void)out_size;

    {   // 0) swizzle/convert inputs into fragment order
        int n = Nn * IN_F;
        swz_x<<<(n + 255) / 256, 256, 0, stream>>>(x, xA);
        int m = HEADS * IN_F * OUT_F;
        swz_w<<<(m + 255) / 256, 256, 0, stream>>>(W, wB);
    }
    // 1) projection (WMMA)
    proj_wmma<<<dim3(Nn / 16, OT_N, HEADS), 32, 0, stream>>>(xA, wB, h16, hB);
    // 2) attention vectors (pre-scaled by log2e)
    {
        int n = HEADS * Nn;
        attvec<<<(n + 255) / 256, 256, 0, stream>>>(h16, a_i, a_j, ai, aj);
    }
    // 3) flash attention (WMMA) -> y pre-BN
    flash_att<<<Nn / 32, 256, 0, stream>>>(graph, ai, aj, hB, y);
    // 4) BN stats
    bn_stats<<<CH, 256, 0, stream>>>(y, mu, rs);
    // 5) BN apply + ReLU
    {
        int n = Nn * CH;
        bn_apply<<<(n + 255) / 256, 256, 0, stream>>>(y, mu, rs, gamma, beta);
    }
}
